// SlidingWindowAttention_21079699488983
// MI455X (gfx1250) — matrix operand order repaired and checked by reading, not compiled, not run on hardware
//
#include <hip/hip_runtime.h>
#include <stdint.h>

typedef __attribute__((ext_vector_type(16))) _Float16 v16h;
typedef __attribute__((ext_vector_type(8)))  _Float16 vh8;
typedef __attribute__((ext_vector_type(4)))  _Float16 h4;
typedef __attribute__((ext_vector_type(2)))  __fp16   fp16x2;  // cvt_pkrtz result type
typedef __attribute__((ext_vector_type(4)))  float    vf4;
typedef __attribute__((ext_vector_type(8)))  float    v8f;

#define TILE_M 128
#define TILE_N 128
#define TILE_K 32
#define LDSP   40   // pitch in halves: 80B stride -> 20-bank step, rows 16B-aligned for b128
#define PFD    4    // prefetch distance in K-chunks for the HBM-resident A stream

union Frag { uint32_t u[8]; v16h v; };

// ---- one-time per call: Wt[n][k] (f16) = W[k][n] (f32), N x N ----
__global__ __launch_bounds__(256)
void transpose_w_f16(const float* __restrict__ W, _Float16* __restrict__ Wt, int N)
{
    __shared__ _Float16 t[32][33];
    const int tid = threadIdx.x;
    const int bn  = blockIdx.x * 32;   // n tile
    const int bk  = blockIdx.y * 32;   // k tile
    const int c = tid & 31, r0 = tid >> 5;  // 8 rows per pass
#pragma unroll
    for (int p = 0; p < 4; ++p) {
        int r = r0 + p * 8;
        t[r][c] = (_Float16)W[(long long)(bk + r) * N + (bn + c)];  // coalesced along n
    }
    __syncthreads();
#pragma unroll
    for (int p = 0; p < 4; ++p) {
        int r = r0 + p * 8;
        Wt[(long long)(bn + r) * N + (bk + c)] = t[c][r];           // coalesced along k
    }
}

// C[M,N] (f16) = A[M,K] (f32, row stride lda) * Wt^T + bias ; Wt is [N][K] f16 row-major
__global__ __launch_bounds__(256)
void proj_gemm_f16(const float* __restrict__ A, long long lda,
                   const _Float16* __restrict__ Wt,
                   const float* __restrict__ bias,
                   _Float16* __restrict__ C, int M, int N, int K)
{
    __shared__ _Float16 sA[2][TILE_M * LDSP];
    __shared__ _Float16 sB[2][TILE_N * LDSP];   // sB[buf][n][k]

    const int tid   = threadIdx.x;
    const int lane  = tid & 31;
    const int wid   = tid >> 5;            // 0..7
    const int waveM = (wid >> 1) * 32;     // 0,32,64,96
    const int waveN = (wid & 1) * 64;      // 0,64

    const int m0 = blockIdx.y * TILE_M;
    const int n0 = blockIdx.x * TILE_N;

    const int r  = lane & 15;
    const int hi = lane >> 4;

    // ---- hoisted staging geometry ----
    // A: 4 x float4 per thread; 8 float4 per 32-elem row
    const float* aptr[4]; int amm[4], akk[4]; bool aok[4];
#pragma unroll
    for (int i = 0; i < 4; ++i) {
        int idx4 = tid + i * 256;
        amm[i] = idx4 >> 3;
        akk[i] = (idx4 & 7) * 4;
        int gm  = m0 + amm[i];
        aok[i]  = gm < M;
        int gmc = aok[i] ? gm : 0;
        aptr[i] = A + (long long)gmc * lda + akk[i];
    }
    // B: 2 x (8 halves) per thread; 4 groups of 8 halves per 32-k row
    const _Float16* btp[2]; int bnn[2], bkk[2];
#pragma unroll
    for (int i = 0; i < 2; ++i) {
        int idx8 = tid + i * 256;
        bnn[i] = idx8 >> 2;
        bkk[i] = (idx8 & 3) * 8;
        btp[i] = Wt + (long long)(n0 + bnn[i]) * K + bkk[i];
    }

    v8f acc[2][4];
#pragma unroll
    for (int i = 0; i < 2; ++i)
#pragma unroll
        for (int j = 0; j < 4; ++j) acc[i][j] = (v8f)0.0f;

    vf4 aR[4]; vh8 bR[2];

    // ---- prologue: stage chunk 0, store to buffer 0 ----
#pragma unroll
    for (int i = 0; i < 4; ++i) {
        vf4 t = *(const vf4*)(aptr[i]);
        aR[i] = aok[i] ? t : (vf4)0.0f;
    }
#pragma unroll
    for (int i = 0; i < 2; ++i) bR[i] = *(const vh8*)(btp[i]);
#pragma unroll
    for (int i = 0; i < 4; ++i) {
        fp16x2 p0 = __builtin_amdgcn_cvt_pkrtz(aR[i][0], aR[i][1]);
        fp16x2 p1 = __builtin_amdgcn_cvt_pkrtz(aR[i][2], aR[i][3]);
        uint2 pk;
        pk.x = __builtin_bit_cast(uint32_t, p0);
        pk.y = __builtin_bit_cast(uint32_t, p1);
        *(uint2*)&sA[0][amm[i] * LDSP + akk[i]] = pk;
    }
#pragma unroll
    for (int i = 0; i < 2; ++i)
        *(vh8*)&sB[0][bnn[i] * LDSP + bkk[i]] = bR[i];
    __syncthreads();

    const int NCH = K / TILE_K;
    for (int kc = 0; kc < NCH; ++kc) {
        const int cur = kc & 1;
        const bool more = (kc + 1) < NCH;

        // ---- prefetch A stream PFD chunks ahead (global_prefetch_b8; speculative) ----
        if (kc + PFD < NCH) {
            const long long kp = (long long)(kc + PFD) * TILE_K;
#pragma unroll
            for (int i = 0; i < 4; ++i)
                __builtin_prefetch(aptr[i] + kp, 0, 3);
        }

        // ---- issue next chunk's global loads (overlap with ds + wmma) ----
        if (more) {
            const long long ka = (long long)(kc + 1) * TILE_K;
#pragma unroll
            for (int i = 0; i < 4; ++i) {
                vf4 t = *(const vf4*)(aptr[i] + ka);
                aR[i] = aok[i] ? t : (vf4)0.0f;
            }
#pragma unroll
            for (int i = 0; i < 2; ++i) bR[i] = *(const vh8*)(btp[i] + ka);
        }

        // ---- fragments from current buffer (ISA 7.12.2 16-bit layouts) ----
        Frag af[2], bf[4];
#pragma unroll
        for (int mi = 0; mi < 2; ++mi) {
            const _Float16* base = &sA[cur][(waveM + mi * 16 + r) * LDSP];
#pragma unroll
            for (int v = 0; v < 4; ++v)
                af[mi].u[v]     = *(const uint32_t*)&base[hi * 8 + 2 * v];
#pragma unroll
            for (int v = 0; v < 4; ++v)
                af[mi].u[4 + v] = *(const uint32_t*)&base[16 + hi * 8 + 2 * v];
        }
#pragma unroll
        for (int ni = 0; ni < 4; ++ni) {
            const _Float16* base = &sB[cur][(waveN + ni * 16 + r) * LDSP];
#pragma unroll
            for (int v = 0; v < 8; ++v)
                bf[ni].u[v] = *(const uint32_t*)&base[(v >> 2) * 16 + hi * 8 + 2 * (v & 3)];
        }

#pragma unroll
        for (int mi = 0; mi < 2; ++mi)
#pragma unroll
            for (int ni = 0; ni < 4; ++ni)
                acc[mi][ni] = __builtin_amdgcn_wmma_f32_16x16x32_f16(
                    false, af[mi].v, false, bf[ni].v,
                    (short)0, acc[mi][ni], false, false);

        if (more) {
            const int nxt = cur ^ 1;
            __syncthreads();   // all waves finished reading buffer nxt one iter ago
#pragma unroll
            for (int i = 0; i < 4; ++i) {
                fp16x2 p0 = __builtin_amdgcn_cvt_pkrtz(aR[i][0], aR[i][1]);
                fp16x2 p1 = __builtin_amdgcn_cvt_pkrtz(aR[i][2], aR[i][3]);
                uint2 pk;
                pk.x = __builtin_bit_cast(uint32_t, p0);
                pk.y = __builtin_bit_cast(uint32_t, p1);
                *(uint2*)&sA[nxt][amm[i] * LDSP + akk[i]] = pk;
            }
#pragma unroll
            for (int i = 0; i < 2; ++i)
                *(vh8*)&sB[nxt][bnn[i] * LDSP + bkk[i]] = bR[i];
            __syncthreads();   // stores visible before next iteration's reads
        }
    }

    // ---- epilogue: +bias, f32 -> f16 store ----
#pragma unroll
    for (int mi = 0; mi < 2; ++mi) {
#pragma unroll
        for (int ni = 0; ni < 4; ++ni) {
            int   ncol = n0 + waveN + ni * 16 + r;
            float bv   = bias[ncol];
#pragma unroll
            for (int rr = 0; rr < 8; ++rr) {
                int grow = m0 + waveM + mi * 16 + rr + hi * 8;
                if (grow < M)
                    C[(long long)grow * N + ncol] = (_Float16)(acc[mi][ni][rr] + bv);
            }
        }
    }
}

// One workgroup per (b, s): windowed scores -> softmax -> weighted V sum.
__global__ __launch_bounds__(256)
void swa_attend(const _Float16* __restrict__ Qp,  // [B*S][H]
                const _Float16* __restrict__ Kp,  // [B*T][H]
                const _Float16* __restrict__ Vp,  // [B*T][H]
                float* __restrict__ Out)          // [B*S][H]
{
    const int Tn = 4096, Hn = 2048, Wn = 256;
    const int bs  = blockIdx.x;       // 0..63
    const int b   = bs >> 4;
    const int s   = bs & 15;
    const int tid = threadIdx.x;

    __shared__ float qs[2048];
    __shared__ float sc[Wn + 1];
    __shared__ float redv[1];

    const _Float16* qrow = Qp + (long long)bs * Hn;
    for (int i = tid; i < Hn; i += 256) qs[i] = (float)qrow[i];
    __syncthreads();

    const int   center = s * Wn - Wn / 2;          // pos - 128
    const float scale  = rsqrtf((float)Hn);

    // scores (j = 0..256): thread-per-j dot products
    for (int j0 = 0; j0 <= Wn; j0 += 256) {
        int j = j0 + tid;
        if (j <= Wn) {
            int row = center + j;
            if (row >= 0 && row < Tn) {
                const h4* kr = (const h4*)(Kp + ((long long)b * Tn + row) * Hn);
                float a0 = 0.0f, a1 = 0.0f, a2 = 0.0f, a3 = 0.0f;
                for (int h = 0; h < Hn / 4; ++h) {
                    h4 kv = kr[h];
                    a0 += (float)kv[0] * qs[4 * h + 0];
                    a1 += (float)kv[1] * qs[4 * h + 1];
                    a2 += (float)kv[2] * qs[4 * h + 2];
                    a3 += (float)kv[3] * qs[4 * h + 3];
                }
                sc[j] = (a0 + a1 + a2 + a3) * scale;
            } else {
                sc[j] = -INFINITY;
            }
        }
    }
    __syncthreads();

    // tiny serial softmax normalization (257 elems; negligible vs memory time)
    if (tid == 0) {
        float m = -INFINITY;
        for (int j = 0; j <= Wn; ++j) m = fmaxf(m, sc[j]);
        float sum = 0.0f;
        for (int j = 0; j <= Wn; ++j) { float e = __expf(sc[j] - m); sc[j] = e; sum += e; }
        redv[0] = 1.0f / sum;
    }
    __syncthreads();
    const float inv = redv[0];

    // output: thread owns 8 h-values; fully coalesced V row streaming
    const int h0 = tid * 8;
    float o[8];
#pragma unroll
    for (int i = 0; i < 8; ++i) o[i] = 0.0f;

    for (int j = 0; j <= Wn; ++j) {
        int row = center + j;
        if (row < 0 || row >= Tn) continue;       // uniform across block
        float wj = sc[j] * inv;
        const h4* vr = (const h4*)(Vp + ((long long)b * Tn + row) * Hn + h0);
        h4 v0 = vr[0], v1 = vr[1];
        o[0] += wj * (float)v0[0]; o[1] += wj * (float)v0[1];
        o[2] += wj * (float)v0[2]; o[3] += wj * (float)v0[3];
        o[4] += wj * (float)v1[0]; o[5] += wj * (float)v1[1];
        o[6] += wj * (float)v1[2]; o[7] += wj * (float)v1[3];
    }

    float* orow = Out + (long long)bs * Hn + h0;
#pragma unroll
    for (int i = 0; i < 8; ++i) orow[i] = o[i];
}

extern "C" void kernel_launch(void* const* d_in, const int* in_sizes, int n_in,
                              void* d_out, int out_size, void* d_ws, size_t ws_size,
                              hipStream_t stream) {
    const float* q  = (const float*)d_in[0];
    const float* k  = (const float*)d_in[1];
    const float* v  = (const float*)d_in[2];
    const float* Wq = (const float*)d_in[3];
    const float* bq = (const float*)d_in[4];
    const float* Wk = (const float*)d_in[5];
    const float* bk = (const float*)d_in[6];
    const float* Wv = (const float*)d_in[7];
    const float* bv = (const float*)d_in[8];
    float* out = (float*)d_out;

    const int B = 4, T = 4096, H = 2048, S = 16;
    const int MKV = B * T;   // 16384
    const int MQ  = B * S;   // 64

    _Float16* Kp  = (_Float16*)d_ws;
    _Float16* Vp  = Kp  + (size_t)MKV * H;
    _Float16* Qp  = Vp  + (size_t)MKV * H;
    _Float16* WtQ = Qp  + (size_t)MQ * H;
    _Float16* WtK = WtQ + (size_t)H * H;
    _Float16* WtV = WtK + (size_t)H * H;   // total ~160 MB of workspace

    dim3 blk(256);

    // one-time weight transpose + f32->f16 (3 x ~8MB outputs)
    dim3 gT(H / 32, H / 32);               // (64,64)
    hipLaunchKernelGGL(transpose_w_f16, gT, blk, 0, stream, Wq, WtQ, H);
    hipLaunchKernelGGL(transpose_w_f16, gT, blk, 0, stream, Wk, WtK, H);
    hipLaunchKernelGGL(transpose_w_f16, gT, blk, 0, stream, Wv, WtV, H);

    dim3 gKV(H / TILE_N, MKV / TILE_M);    // (16, 128)
    hipLaunchKernelGGL(proj_gemm_f16, gKV, blk, 0, stream,
                       k, (long long)H, WtK, bk, Kp, MKV, H, H);
    hipLaunchKernelGGL(proj_gemm_f16, gKV, blk, 0, stream,
                       v, (long long)H, WtV, bv, Vp, MKV, H, H);

    // Q: gathered rows are every 256th row of flat q -> same GEMM with lda = 256*H
    dim3 gQ(H / TILE_N, 1);
    hipLaunchKernelGGL(proj_gemm_f16, gQ, blk, 0, stream,
                       q, (long long)(256 * H), WtQ, bq, Qp, MQ, H, H);

    hipLaunchKernelGGL(swa_attend, dim3(B * S), blk, 0, stream, Qp, Kp, Vp, out);
}
